// BiLSTM_9749575762682
// MI455X (gfx1250) — compile-verified, matrix operand order checked by reading, not run on hardware
//
#include <hip/hip_runtime.h>
#include <cstdint>
#include <cstddef>

// ---------------- CDNA5 WMMA helpers (wave32, gfx1250) ----------------
typedef __attribute__((ext_vector_type(16))) __bf16 v16bf;
typedef __attribute__((ext_vector_type(8)))  float  v8f;

struct Q2 { uint4 lo, hi; };
static_assert(sizeof(Q2) == 32, "frag size");

__device__ __forceinline__ v16bf make_frag(uint4 lo, uint4 hi) {
  Q2 q{lo, hi};
  return __builtin_bit_cast(v16bf, q);
}

__device__ __forceinline__ v8f wmma_bf16(v16bf a, v16bf b, v8f c) {
  // D = A(16x32 bf16) * B(32x16 bf16) + C(16x16 f32)
  return __builtin_amdgcn_wmma_f32_16x16x32_bf16(false, a, false, b, (short)0, c,
                                                 false, false);
}

__device__ __forceinline__ unsigned short f2bf(float f) {
  unsigned int u = __float_as_uint(f);
  u += 0x7FFFu + ((u >> 16) & 1u);          // round-to-nearest-even
  return (unsigned short)(u >> 16);
}

__device__ __forceinline__ float sigmoidf(float x) { return 1.0f / (1.0f + __expf(-x)); }

#if __has_builtin(__builtin_amdgcn_global_load_async_to_lds_b128)
#define HAVE_ASYNC_LDS 1
// Builtin signature: (int __vector(4) AS1*, int __vector(4) AS3*, Imm offset, Imm cpol)
typedef int v4i_t __attribute__((vector_size(16)));
typedef __attribute__((address_space(1))) v4i_t* gv4p;
typedef __attribute__((address_space(3))) v4i_t* lv4p;
#endif

// ---------------- Problem constants ----------------
#define TT    128
#define BB    64
#define WW    16
#define EMB   300
#define CEMB  32
#define CCH   128
#define HID   512
#define G4H   2048
#define KIN   428
#define KINP  448          // padded to 14*32
#define OUTN  20
#define KFC   1024
#define ROWS  8192         // T*B

// tiled weight element strides (ushort counts)
#define WIH_DIR (128 * 14 * 32 * 16)   // 917504
#define WHH_DIR (128 * 16 * 32 * 16)   // 1048576

// ---------------- workspace layout (bytes) ----------------
static const size_t OFS_WIH  = 0;                       // bf16 tiled  2*WIH_DIR*2
static const size_t OFS_WHH  = OFS_WIH + 3670016;       // bf16 tiled  2*WHH_DIR*2
static const size_t OFS_FCW  = OFS_WHH + 4194304;       // bf16 tiled  2*32*32*16*2 = 65536
static const size_t OFS_BIAS = OFS_FCW + 65536;         // f32 [2][2048]
static const size_t OFS_FEAT = OFS_BIAS + 16384;        // bf16 [8192][448]
static const size_t OFS_XW   = OFS_FEAT + 7340032;      // f32 [2][8192][2048]
static const size_t OFS_H    = OFS_XW + 134217728;      // bf16 [2][64][512]
static const size_t OFS_C    = OFS_H + 131072;          // f32  [2][64][512]
static const size_t OFS_HS   = OFS_C + 262144;          // bf16 [T][64][1024]

// ---------------- kernels ----------------
__global__ void zero_kernel(unsigned int* p, int n) {
  int i = blockIdx.x * 256 + threadIdx.x;
  if (i < n) p[i] = 0u;
}

// Pre-swizzle W (nrows x kcols, row-major f32) into WMMA B-fragment order:
// dst[((nt*ktiles + kt)*32 + lane)*16 + j] holds B(k,n) = W[n][k] as bf16,
// with n = nt*16 + lane%16 and k = kt*32 + (lane/16)*8 + (j<8 ? j : j+8).
__global__ void tile_weight(const float* __restrict__ src, int nrows, int kcols,
                            int ktiles, unsigned short* __restrict__ dst) {
  int nt = blockIdx.x, kt = blockIdx.y, lane = threadIdx.x;   // 32 threads
  int n = nt * 16 + (lane & 15);
  int khalf = (lane >> 4) * 8;
  unsigned short vals[16];
#pragma unroll
  for (int j = 0; j < 16; ++j) {
    int k = kt * 32 + khalf + (j < 8 ? j : j + 8);
    float f = (n < nrows && k < kcols) ? src[(size_t)n * kcols + k] : 0.0f;
    vals[j] = f2bf(f);
  }
  unsigned long long* d =
      (unsigned long long*)(dst + ((size_t)(nt * ktiles + kt) * 32 + lane) * 16);
  const unsigned long long* s = (const unsigned long long*)vals;
#pragma unroll
  for (int q = 0; q < 4; ++q) d[q] = s[q];
}

__global__ void bias_kernel(const float* __restrict__ bihf, const float* __restrict__ bhhf,
                            const float* __restrict__ bihb, const float* __restrict__ bhhb,
                            float* __restrict__ bias) {
  int i = blockIdx.x * 256 + threadIdx.x;        // 0..2047
  if (i < G4H) {
    bias[i]        = bihf[i] + bhhf[i];
    bias[G4H + i]  = bihb[i] + bhhb[i];
  }
}

// word embedding gather + depthwise char conv (groups=32, 4 filters/group) + maxpool
__global__ void build_feat(const int* __restrict__ words, const int* __restrict__ chars,
                           const float* __restrict__ embW, const float* __restrict__ cembW,
                           const float* __restrict__ convW, const float* __restrict__ convB,
                           unsigned short* __restrict__ feat) {
  int row = blockIdx.x;                 // t*B + b
  int t = row >> 6, b = row & 63;
  int tid = threadIdx.x;                // 128 threads
  __shared__ float ce[WW][CEMB];
  for (int i = tid; i < WW * CEMB; i += 128) {
    int w = i >> 5, c = i & 31;
    int cid = chars[((size_t)b * TT + t) * WW + w];
    ce[w][c] = cembW[(size_t)cid * CEMB + c];
  }
  __syncthreads();
  unsigned short* frow = feat + (size_t)row * KINP;
  int wid = words[row];
  for (int j = tid; j < EMB; j += 128)
    frow[j] = f2bf(embW[(size_t)wid * EMB + j]);
  {
    int ch = tid;                       // 0..127
    int g = ch >> 2;                    // input channel for this group
    float w0 = convW[ch * 3 + 0], w1 = convW[ch * 3 + 1], w2 = convW[ch * 3 + 2];
    float m = -3.4e38f;
#pragma unroll
    for (int p = 0; p < WW - 2; ++p) {
      float s = ce[p][g] * w0 + ce[p + 1][g] * w1 + ce[p + 2][g] * w2;
      m = fmaxf(m, s);
    }
    frow[EMB + ch] = f2bf(m + convB[ch]);
  }
  if (tid < KINP - KIN) frow[KIN + tid] = 0;   // zero K padding
}

// xW[d] = feat(8192x448 bf16) * WihT[d] + bias[d]   -> f32 [2][8192][2048]
__global__ void gemm_xw(const unsigned short* __restrict__ feat,
                        const unsigned short* __restrict__ WihT,
                        const float* __restrict__ bias, float* __restrict__ xW) {
  int d = blockIdx.z;
  int rowBase = blockIdx.x * 64;
  int nBase = blockIdx.y * 64;
  int wave = threadIdx.x >> 5, lane = threadIdx.x & 31;
  int msub = wave & 3, nsub = wave >> 2;       // 4x2 wave grid over 64x64 tile
  int half = lane >> 4, mloc = lane & 15, ncol = lane & 15;
  const unsigned short* WT = WihT + (size_t)d * WIH_DIR;
  int row = rowBase + msub * 16 + mloc;
  const unsigned short* arow = feat + (size_t)row * KINP;
  float b0 = bias[d * G4H + nBase + nsub * 32 + ncol];
  float b1 = bias[d * G4H + nBase + nsub * 32 + 16 + ncol];
  v8f acc0, acc1;
#pragma unroll
  for (int v = 0; v < 8; ++v) { acc0[v] = b0; acc1[v] = b1; }
  int nt0 = (nBase >> 4) + nsub * 2;
  int nt1 = nt0 + 1;
  for (int kt = 0; kt < 14; ++kt) {
    int kofs = kt * 32 + half * 8;
    v16bf a = make_frag(*(const uint4*)(arow + kofs),
                        *(const uint4*)(arow + kofs + 16));
    const uint4* bp0 = (const uint4*)(WT + ((size_t)(nt0 * 14 + kt) * 32 + lane) * 16);
    const uint4* bp1 = (const uint4*)(WT + ((size_t)(nt1 * 14 + kt) * 32 + lane) * 16);
    if (kt + 1 < 14) {
      __builtin_prefetch(WT + ((size_t)(nt0 * 14 + kt + 1) * 32 + lane) * 16, 0, 1);
      __builtin_prefetch(WT + ((size_t)(nt1 * 14 + kt + 1) * 32 + lane) * 16, 0, 1);
    }
    acc0 = wmma_bf16(a, make_frag(bp0[0], bp0[1]), acc0);
    acc1 = wmma_bf16(a, make_frag(bp1[0], bp1[1]), acc1);
  }
  float* orow = xW + ((size_t)d * ROWS + rowBase) * G4H;
#pragma unroll
  for (int v = 0; v < 8; ++v) {
    int r = msub * 16 + v + 8 * half;
    orow[(size_t)r * G4H + nBase + nsub * 32 + ncol]      = acc0[v];
    orow[(size_t)r * G4H + nBase + nsub * 32 + 16 + ncol] = acc1[v];
  }
}

// One recurrent step, both directions. Block = 512 thr (16 waves) handles 16 batch
// rows x all 512 hidden cols; each wave owns 2 hidden col-tiles and, for each,
// computes all 4 gate tiles (i,f,g,o) so the cell update stays in registers.
__global__ void lstm_step(int t, const float* __restrict__ xW,
                          const unsigned short* __restrict__ WhhT,
                          unsigned short* __restrict__ h_state,
                          float* __restrict__ c_state,
                          unsigned short* __restrict__ hs) {
  int rg = blockIdx.x;                  // 0..3 row group (16 batch rows)
  int d = blockIdx.y;                   // direction
  int mBase = rg * 16;
  int tid = threadIdx.x;                // 512
  __shared__ unsigned short hsm[16 * HID];
  const unsigned short* hsrc = h_state + ((size_t)d * BB + mBase) * HID;
#ifdef HAVE_ASYNC_LDS
  // gfx1250 async global->LDS staging (tracked by ASYNCcnt, no VGPR round-trip)
  for (int i = tid; i < 16 * HID / 8; i += 512) {
    gv4p g = (gv4p)(unsigned long long)(size_t)(((const uint4*)hsrc) + i);
    lv4p l = (lv4p)(unsigned)(size_t)(((uint4*)hsm) + i);
    __builtin_amdgcn_global_load_async_to_lds_b128(g, l, 0, 0);
  }
#if __has_builtin(__builtin_amdgcn_s_wait_asynccnt)
  __builtin_amdgcn_s_wait_asynccnt(0);
#else
  asm volatile("s_wait_asynccnt 0x0" ::: "memory");
#endif
#else
  for (int i = tid; i < 16 * HID / 8; i += 512)
    ((uint4*)hsm)[i] = ((const uint4*)hsrc)[i];
#endif
  __syncthreads();
  int wave = tid >> 5, lane = tid & 31;
  int half = lane >> 4, mloc = lane & 15, ncol = lane & 15;
  int tx = d ? (TT - 1 - t) : t;
  const float* xrow = xW + ((size_t)d * ROWS + (size_t)tx * BB + mBase) * G4H;
  const unsigned short* WT = WhhT + (size_t)d * WHH_DIR;
#pragma unroll
  for (int cti = 0; cti < 2; ++cti) {
    int ct = wave + cti * 16;           // hidden col tile 0..31
    v8f ai, af, ag, ao;
#pragma unroll
    for (int v = 0; v < 8; ++v) {
      size_t ro = (size_t)(v + 8 * half) * G4H + ct * 16 + ncol;
      ai[v] = xrow[ro];
      af[v] = xrow[ro + 512];
      ag[v] = xrow[ro + 1024];
      ao[v] = xrow[ro + 1536];
    }
    for (int kt = 0; kt < 16; ++kt) {
      int kofs = kt * 32 + half * 8;
      const unsigned short* ap = hsm + (size_t)mloc * HID + kofs;
      v16bf a = make_frag(*(const uint4*)ap, *(const uint4*)(ap + 16));
      const uint4* bi = (const uint4*)(WT + ((size_t)((ct)*16 + kt) * 32 + lane) * 16);
      const uint4* bf_ = (const uint4*)(WT + ((size_t)((32 + ct) * 16 + kt) * 32 + lane) * 16);
      const uint4* bg = (const uint4*)(WT + ((size_t)((64 + ct) * 16 + kt) * 32 + lane) * 16);
      const uint4* bo = (const uint4*)(WT + ((size_t)((96 + ct) * 16 + kt) * 32 + lane) * 16);
      if (kt + 1 < 16)
        __builtin_prefetch(WT + ((size_t)((ct)*16 + kt + 1) * 32 + lane) * 16, 0, 1);
      ai = wmma_bf16(a, make_frag(bi[0], bi[1]), ai);
      af = wmma_bf16(a, make_frag(bf_[0], bf_[1]), af);
      ag = wmma_bf16(a, make_frag(bg[0], bg[1]), ag);
      ao = wmma_bf16(a, make_frag(bo[0], bo[1]), ao);
    }
    // LSTM cell update, in registers
#pragma unroll
    for (int v = 0; v < 8; ++v) {
      int r = mBase + v + 8 * half;
      int hc = ct * 16 + ncol;
      size_t ci = ((size_t)d * BB + r) * HID + hc;
      float cprev = c_state[ci];
      float gi = sigmoidf(ai[v]);
      float gf = sigmoidf(af[v]);
      float gg = tanhf(ag[v]);
      float go = sigmoidf(ao[v]);
      float cn = gf * cprev + gi * gg;
      float hn = go * tanhf(cn);
      c_state[ci] = cn;
      unsigned short hb = f2bf(hn);
      h_state[ci] = hb;
      hs[((size_t)tx * BB + r) * KFC + d * HID + hc] = hb;
    }
  }
}

// out[row, 0..19] = concat(h_f,h_b)[row,1024] * fcW^T + fc_b
__global__ void fc_kernel(const unsigned short* __restrict__ hs,
                          const unsigned short* __restrict__ fcWT,
                          const float* __restrict__ fcb, float* __restrict__ out) {
  int rowBase = blockIdx.x * 16;
  int wave = threadIdx.x >> 5;          // 2 waves; wave == n-tile (cols 0..15 / 16..31)
  int lane = threadIdx.x & 31;
  int half = lane >> 4, mloc = lane & 15, ncol = lane & 15;
  int col = wave * 16 + ncol;
  float b = (col < OUTN) ? fcb[col] : 0.0f;
  v8f acc;
#pragma unroll
  for (int v = 0; v < 8; ++v) acc[v] = b;
  const unsigned short* arow = hs + (size_t)(rowBase + mloc) * KFC;
  for (int kt = 0; kt < 32; ++kt) {
    int kofs = kt * 32 + half * 8;
    v16bf a = make_frag(*(const uint4*)(arow + kofs),
                        *(const uint4*)(arow + kofs + 16));
    const uint4* bp = (const uint4*)(fcWT + ((size_t)(wave * 32 + kt) * 32 + lane) * 16);
    acc = wmma_bf16(a, make_frag(bp[0], bp[1]), acc);
  }
  if (col < OUTN)
#pragma unroll
    for (int v = 0; v < 8; ++v)
      out[(size_t)(rowBase + v + 8 * half) * OUTN + col] = acc[v];
}

// ---------------- launcher ----------------
extern "C" void kernel_launch(void* const* d_in, const int* in_sizes, int n_in,
                              void* d_out, int out_size, void* d_ws, size_t ws_size,
                              hipStream_t stream) {
  const int*   words  = (const int*)d_in[0];
  const int*   chars  = (const int*)d_in[1];
  const float* embW   = (const float*)d_in[2];
  const float* cembW  = (const float*)d_in[3];
  const float* convW  = (const float*)d_in[4];
  const float* convB  = (const float*)d_in[5];
  const float* Wih_f  = (const float*)d_in[6];
  const float* Whh_f  = (const float*)d_in[7];
  const float* bih_f  = (const float*)d_in[8];
  const float* bhh_f  = (const float*)d_in[9];
  const float* Wih_b  = (const float*)d_in[10];
  const float* Whh_b  = (const float*)d_in[11];
  const float* bih_b  = (const float*)d_in[12];
  const float* bhh_b  = (const float*)d_in[13];
  const float* fcW    = (const float*)d_in[14];
  const float* fcb    = (const float*)d_in[15];
  float* out = (float*)d_out;

  char* ws = (char*)d_ws;
  unsigned short* WihT  = (unsigned short*)(ws + OFS_WIH);
  unsigned short* WhhT  = (unsigned short*)(ws + OFS_WHH);
  unsigned short* fcWT  = (unsigned short*)(ws + OFS_FCW);
  float*          bias  = (float*)(ws + OFS_BIAS);
  unsigned short* feat  = (unsigned short*)(ws + OFS_FEAT);
  float*          xW    = (float*)(ws + OFS_XW);
  unsigned short* hstate = (unsigned short*)(ws + OFS_H);
  float*          cstate = (float*)(ws + OFS_C);
  unsigned short* hs    = (unsigned short*)(ws + OFS_HS);

  // zero h/c state (contiguous region: 131072 + 262144 bytes = 98304 uints)
  zero_kernel<<<dim3(384), dim3(256), 0, stream>>>((unsigned int*)(ws + OFS_H), 98304);

  // weight swizzle to WMMA B-fragment order (bf16)
  tile_weight<<<dim3(128, 14), dim3(32), 0, stream>>>(Wih_f, G4H, KIN, 14, WihT);
  tile_weight<<<dim3(128, 14), dim3(32), 0, stream>>>(Wih_b, G4H, KIN, 14, WihT + WIH_DIR);
  tile_weight<<<dim3(128, 16), dim3(32), 0, stream>>>(Whh_f, G4H, HID, 16, WhhT);
  tile_weight<<<dim3(128, 16), dim3(32), 0, stream>>>(Whh_b, G4H, HID, 16, WhhT + WHH_DIR);
  tile_weight<<<dim3(2, 32),   dim3(32), 0, stream>>>(fcW,  OUTN, KFC, 32, fcWT);

  bias_kernel<<<dim3(8), dim3(256), 0, stream>>>(bih_f, bhh_f, bih_b, bhh_b, bias);

  // features: word emb + char CNN (bf16, K padded 428->448)
  build_feat<<<dim3(ROWS), dim3(128), 0, stream>>>(words, chars, embW, cembW, convW,
                                                   convB, feat);

  // input projections for both directions (big WMMA GEMM)
  gemm_xw<<<dim3(128, 32, 2), dim3(256), 0, stream>>>(feat, WihT, bias, xW);

  // 128 sequential recurrent steps, both directions per launch
  for (int t = 0; t < TT; ++t)
    lstm_step<<<dim3(4, 2), dim3(512), 0, stream>>>(t, xW, WhhT, hstate, cstate, hs);

  // final projection
  fc_kernel<<<dim3(512), dim3(64), 0, stream>>>(hs, fcWT, fcb, out);
}
